// KGramMLPSeqModel_35476429865350
// MI455X (gfx1250) — compile-verified, matrix operand order checked by reading, not run on hardware
//
#include <hip/hip_runtime.h>
#include <hip/hip_bf16.h>

#define SEQ    512
#define BATCH  4
#define KCTX   3
#define VOCAB  32000
#define EMBED  1024
#define MROWS  (SEQ * BATCH)   // 2048 GEMM rows

typedef __attribute__((ext_vector_type(16))) _Float16 v16h;
typedef __attribute__((ext_vector_type(8)))  _Float16 v8h;
typedef __attribute__((ext_vector_type(8)))  float    v8f;

// ---------------------------------------------------------------------------
// Kernel 1: H[m][e] = silu( sum_j W1[j, ctx_j, e] + b1[e] ), stored as f16.
// m = t*BATCH + b ; ctx_j = tokens[t+j-K] (0 if padded).
// ---------------------------------------------------------------------------
__global__ __launch_bounds__(256) void build_h_kernel(
    const int* __restrict__ tokens, const float* __restrict__ W1,
    const float* __restrict__ b1, _Float16* __restrict__ H)
{
  const int m = blockIdx.x;
  const int t = m >> 2;
  const int b = m & 3;

  int tok[KCTX];
#pragma unroll
  for (int j = 0; j < KCTX; ++j) {
    const int tt = t + j - KCTX;
    tok[j] = (tt >= 0) ? tokens[tt * BATCH + b] : 0;
  }

  const size_t VE = (size_t)VOCAB * EMBED;
#pragma unroll
  for (int i = 0; i < EMBED / 256; ++i) {
    const int e = threadIdx.x + i * 256;
    float s = b1[e];
#pragma unroll
    for (int j = 0; j < KCTX; ++j)
      s += W1[(size_t)j * VE + (size_t)tok[j] * EMBED + e];
    s = s / (1.0f + __expf(-s));   // SiLU
    H[(size_t)m * EMBED + e] = (_Float16)s;
  }
}

// ---------------------------------------------------------------------------
// Kernel 2: W2T[v][e] = (f16) W2[e][v]  (LDS-tiled transpose, coalesced both ways)
// ---------------------------------------------------------------------------
__global__ __launch_bounds__(256) void w2_transpose_kernel(
    const float* __restrict__ W2, _Float16* __restrict__ W2T)
{
  __shared__ float tile[32][33];
  const int tx = threadIdx.x & 31;
  const int ty = threadIdx.x >> 5;        // 0..7
  const int v0 = blockIdx.x * 32;
  const int e0 = blockIdx.y * 32;

#pragma unroll
  for (int r = ty; r < 32; r += 8)
    tile[r][tx] = W2[(size_t)(e0 + r) * VOCAB + (v0 + tx)];
  __syncthreads();
#pragma unroll
  for (int r = ty; r < 32; r += 8)
    W2T[(size_t)(v0 + r) * EMBED + (e0 + tx)] = (_Float16)tile[tx][r];
}

// ---------------------------------------------------------------------------
// Kernel 3: out[m][v] = H[m][:] . W2T[v][:] + b2[v]   via v_wmma_f32_16x16x32_f16
//   Block: 256 threads = 8 waves, arranged 2 (M) x 4 (N).
//   Wave tile: 64x64 output = 4x4 WMMA 16x16 tiles, K looped 1024 in steps of 32.
//   A-frag lane map (ISA 7.12.2, 16-bit A 16x32): lane<16 -> K {0..7,16..23},
//   lane>=16 -> K {8..15,24..31}, row = lane&15. B uses the symmetric layout on
//   W2T rows (one output column per lane). D: VGPR r -> row r (+8 if lane>=16).
// ---------------------------------------------------------------------------
#define BN 256
#define BM 128

__global__ __launch_bounds__(256) void gemm_bias_kernel(
    const _Float16* __restrict__ H, const _Float16* __restrict__ W2T,
    const float* __restrict__ b2, float* __restrict__ out)
{
  const int lane = threadIdx.x & 31;
  const int wave = threadIdx.x >> 5;                 // 0..7
  const int m0 = blockIdx.y * BM + (wave >> 2) * 64; // wave row {0,1}
  const int n0 = blockIdx.x * BN + (wave & 3) * 64;  // wave col {0..3}
  const int rc   = lane & 15;                        // A row / B col in tile
  const int koff = (lane & 16) ? 8 : 0;              // K sub-base for this lane

  v8f acc[4][4] = {};

  const _Float16* Aptr[4];
  const _Float16* Bptr[4];
#pragma unroll
  for (int i = 0; i < 4; ++i)
    Aptr[i] = H + (size_t)(m0 + i * 16 + rc) * EMBED + koff;
#pragma unroll
  for (int j = 0; j < 4; ++j)
    Bptr[j] = W2T + (size_t)(n0 + j * 16 + rc) * EMBED + koff;

  for (int k = 0; k < EMBED; k += 32) {
    v16h a[4], bb[4];
#pragma unroll
    for (int i = 0; i < 4; ++i) {
      v8h lo = *(const v8h*)(Aptr[i] + k);
      v8h hi = *(const v8h*)(Aptr[i] + k + 16);
      a[i] = __builtin_shufflevector(lo, hi, 0,1,2,3,4,5,6,7,8,9,10,11,12,13,14,15);
    }
#pragma unroll
    for (int j = 0; j < 4; ++j) {
      v8h lo = *(const v8h*)(Bptr[j] + k);
      v8h hi = *(const v8h*)(Bptr[j] + k + 16);
      bb[j] = __builtin_shufflevector(lo, hi, 0,1,2,3,4,5,6,7,8,9,10,11,12,13,14,15);
    }
#pragma unroll
    for (int i = 0; i < 4; ++i)
#pragma unroll
      for (int j = 0; j < 4; ++j)
        acc[i][j] = __builtin_amdgcn_wmma_f32_16x16x32_f16(
            false, a[i], false, bb[j], (short)0, acc[i][j], false, false);
  }

  const int rsel = (lane & 16) ? 8 : 0;
#pragma unroll
  for (int j = 0; j < 4; ++j) {
    const int col   = n0 + j * 16 + rc;
    const float bias = b2[col];
#pragma unroll
    for (int i = 0; i < 4; ++i) {
      const size_t rowBase = (size_t)(m0 + i * 16 + rsel);
#pragma unroll
      for (int r = 0; r < 8; ++r)
        out[(rowBase + r) * (size_t)VOCAB + col] = acc[i][j][r] + bias;
    }
  }
}

// ---------------------------------------------------------------------------
extern "C" void kernel_launch(void* const* d_in, const int* in_sizes, int n_in,
                              void* d_out, int out_size, void* d_ws, size_t ws_size,
                              hipStream_t stream) {
  const int*   tokens = (const int*)  d_in[0];
  const float* W1     = (const float*)d_in[1];
  const float* b1     = (const float*)d_in[2];
  const float* W2     = (const float*)d_in[3];
  const float* b2     = (const float*)d_in[4];
  float* out = (float*)d_out;

  // workspace layout: [ H : 2048*1024 f16 = 4 MiB ][ W2T : 32000*1024 f16 ]
  _Float16* H   = (_Float16*)d_ws;
  _Float16* W2T = (_Float16*)((char*)d_ws + (size_t)MROWS * EMBED * sizeof(_Float16));

  build_h_kernel<<<MROWS, 256, 0, stream>>>(tokens, W1, b1, H);

  dim3 tgrid(VOCAB / 32, EMBED / 32);
  w2_transpose_kernel<<<tgrid, 256, 0, stream>>>(W2, W2T);

  dim3 ggrid(VOCAB / BN, MROWS / BM);   // 125 x 16
  gemm_bias_kernel<<<ggrid, 256, 0, stream>>>(H, W2T, b2, out);
}
